// MultiHeadAttention_51049981281229
// MI455X (gfx1250) — compile-verified, matrix operand order checked by reading, not run on hardware
//
#include <hip/hip_runtime.h>

// ---- problem constants (match reference) ----
#define kB 2
#define kS 2048
#define kD 1024
#define kH 16
#define kDK 64
#define kDV 64

typedef __attribute__((ext_vector_type(16))) _Float16 v16h;
typedef __attribute__((ext_vector_type(8)))  float    v8f;

#define WMMA_F16(a, b, c) \
    __builtin_amdgcn_wmma_f32_16x16x32_f16(false, (a), false, (b), (short)0, (c), false, false)

// raw v_exp_f32 (args are always <= 0 here; flush-to-zero underflow is desired)
#define EXP2(x) __builtin_amdgcn_exp2f(x)

__device__ __forceinline__ unsigned int pkh(float a, float b) {
    union { _Float16 h[2]; unsigned int u; } p;
    p.h[0] = (_Float16)a; p.h[1] = (_Float16)b;
    return p.u;
}

// CDNA5 async copy: 16B global -> LDS, tracked by ASYNCcnt (no VGPR round-trip).
__device__ __forceinline__ void async_ld16(void* ldsdst, const void* gsrc) {
    unsigned int l = (unsigned int)(size_t)ldsdst;   // flat addr low 32 bits = LDS offset
    asm volatile("global_load_async_to_lds_b128 %0, %1, off"
                 :: "v"(l), "v"(gsrc) : "memory");
}
__device__ __forceinline__ void async_wait0() {
    asm volatile("s_wait_asynccnt 0" ::: "memory");
}

// Load a 16-bit A/B WMMA fragment (16 rows x 32 k) from LDS.
// lanes 0-15: K = {0..7, 16..23}; lanes 16-31: K = {8..15, 24..31}, packed pairs.
__device__ __forceinline__ v16h ld_frag(const _Float16* rowptr, int hi) {
    union { v16h v; unsigned int u[8]; } f;
#pragma unroll
    for (int p = 0; p < 8; ++p) {
        int k = ((p & 3) << 1) + ((p >> 2) << 4) + (hi << 3);
        f.u[p] = *(const unsigned int*)(rowptr + k);
    }
    return f.v;
}

// ============================================================================
// Tiled GEMM: C[M x 1024] = A[M x 1024] * B[1024 x 1024]
// A_HALF:  A is f16 (context, staged via async-to-LDS) else f32 (convert+pack)
// HEAD_OUT: store f16 into [b,h,s,64] head layout, else f32 row-major
// 128x128 block tile, 32 k-step, 256 threads (8 waves), 8 WMMA per wave/iter.
// ============================================================================
template<bool A_HALF, bool HEAD_OUT>
__global__ __launch_bounds__(256)
void gemm_kernel(const void* __restrict__ Aptr, const float* __restrict__ Bw,
                 void* __restrict__ Cptr) {
    constexpr int LD = 40;               // LDS row stride in halves (32 + 8 pad)
    __shared__ _Float16 lA[128 * LD];
    __shared__ _Float16 lB[128 * LD];    // stored transposed: [n][k]

    const int tid  = threadIdx.x;
    const int wave = tid >> 5, lane = tid & 31;
    const int wm   = wave >> 2, wn = wave & 3;   // 2x4 wave grid
    const int hi   = lane >> 4, r  = lane & 15;
    const int m0   = blockIdx.y * 128;
    const int n0   = blockIdx.x * 128;

    v8f acc[4][2];
#pragma unroll
    for (int mt = 0; mt < 4; ++mt)
#pragma unroll
        for (int nt = 0; nt < 2; ++nt)
#pragma unroll
            for (int i = 0; i < 8; ++i) acc[mt][nt][i] = 0.0f;

    for (int k0 = 0; k0 < kD; k0 += 32) {
        __syncthreads();
        // ---- stage A tile 128x32 -> f16 LDS ----
        if (A_HALF) {
            // 128 rows x 4 chunks of 8 halves (16B), async global->LDS
#pragma unroll
            for (int t = 0; t < 2; ++t) {
                int e = tid + t * 256;             // 0..511
                int row = e >> 2, c = e & 3;
                async_ld16(&lA[row * LD + c * 8],
                           (const _Float16*)Aptr + (size_t)(m0 + row) * kD + k0 + c * 8);
            }
        } else {
#pragma unroll
            for (int t = 0; t < 4; ++t) {
                int e   = tid + t * 256;           // 0..1023 quads of 4 floats
                int row = e >> 3, c4 = e & 7;
                const float* g = (const float*)Aptr + (size_t)(m0 + row) * kD + k0 + c4 * 4;
                unsigned int* d = (unsigned int*)&lA[row * LD + c4 * 4];
                d[0] = pkh(g[0], g[1]);
                d[1] = pkh(g[2], g[3]);
            }
        }
        // ---- stage B tile 32x128 (f32) -> transposed f16 LDS [n][k] ----
#pragma unroll
        for (int t = 0; t < 4; ++t) {
            int e  = tid + t * 256;
            int kk = e >> 5, c4 = e & 31;
            const float* g = Bw + (size_t)(k0 + kk) * kD + n0 + c4 * 4;
#pragma unroll
            for (int j = 0; j < 4; ++j)
                lB[(c4 * 4 + j) * LD + kk] = (_Float16)g[j];
        }
        if (A_HALF) async_wait0();
        __syncthreads();

        v16h afr[4], bfr[2];
#pragma unroll
        for (int mt = 0; mt < 4; ++mt)
            afr[mt] = ld_frag(&lA[(wm * 64 + mt * 16 + r) * LD], hi);
#pragma unroll
        for (int nt = 0; nt < 2; ++nt)
            bfr[nt] = ld_frag(&lB[(wn * 32 + nt * 16 + r) * LD], hi);
#pragma unroll
        for (int mt = 0; mt < 4; ++mt)
#pragma unroll
            for (int nt = 0; nt < 2; ++nt)
                acc[mt][nt] = WMMA_F16(afr[mt], bfr[nt], acc[mt][nt]);
    }

    // ---- epilogue ----
#pragma unroll
    for (int mt = 0; mt < 4; ++mt) {
#pragma unroll
        for (int nt = 0; nt < 2; ++nt) {
            int ng = n0 + wn * 32 + nt * 16 + r;
#pragma unroll
            for (int v = 0; v < 8; ++v) {
                int   row = m0 + wm * 64 + mt * 16 + v + 8 * hi;
                float val = acc[mt][nt][v];
                if (HEAD_OUT) {
                    int b = row >> 11, s = row & (kS - 1);
                    int h = ng >> 6,  d = ng & 63;
                    ((_Float16*)Cptr)[((((size_t)b * kH + h) * kS + s) << 6) + d] = (_Float16)val;
                } else {
                    ((float*)Cptr)[(size_t)row * kD + ng] = val;
                }
            }
        }
    }
}

// ============================================================================
// Fused attention: per (b,h,64-query tile). 4 waves x 16 rows.
// Pass 1: QK^T (WMMA) + per-lane online softmax stats, one butterfly merge.
// Pass 2: recompute QK^T, write weights to d_out, stage f16 P in LDS, P*V WMMA.
// Q/K tiles staged with global_load_async_to_lds_b128 (ASYNCcnt).
// ============================================================================
__global__ __launch_bounds__(128)
void attn_kernel(const _Float16* __restrict__ Qw, const _Float16* __restrict__ Kw,
                 const _Float16* __restrict__ Vw, float* __restrict__ Wout,
                 _Float16* __restrict__ ctx) {
    __shared__ _Float16 sQ[64 * 72];       // [qrow][dk]
    __shared__ _Float16 sK[32 * 72];       // [key][dk]
    __shared__ _Float16 sVt[64 * 40];      // [dv][key]  (transposed)
    __shared__ _Float16 sW[4 * 16 * 40];   // per-wave staged P tile [row][key]

    const int tid  = threadIdx.x;
    const int wave = tid >> 5, lane = tid & 31;
    const int hi   = lane >> 4, r  = lane & 15;

    const int qt = blockIdx.x & 31;              // S/64 = 32 query tiles
    const int bh = blockIdx.x >> 5;
    const int h  = bh & (kH - 1), b = bh >> 4;
    const int q0 = qt * 64;

    const _Float16* Qh = Qw + (((size_t)b * kH + h) * kS) * 64;
    const _Float16* Kh = Kw + (((size_t)b * kH + h) * kS) * 64;
    const _Float16* Vh = Vw + (((size_t)b * kH + h) * kS) * 64;

    // ---- load Q tile 64x64: 64 rows x 8 chunks of 8 halves, async ----
#pragma unroll
    for (int t = 0; t < 4; ++t) {
        int e = tid + t * 128;                   // 0..511
        int row = e >> 3, c = e & 7;
        async_ld16(&sQ[row * 72 + c * 8], Qh + (size_t)(q0 + row) * 64 + c * 8);
    }
    async_wait0();
    __syncthreads();

    // persistent Q fragments (16 rows x dk 0..31 / 32..63)
    const _Float16* qrow = &sQ[(wave * 16 + r) * 72];
    v16h qa0 = ld_frag(qrow, hi);
    v16h qa1 = ld_frag(qrow + 32, hi);

    const float LOG2E  = 1.4426950408889634f;
    const float scale2 = LOG2E * 0.125f;                   // 1/sqrt(64), log2 domain
    const float slope2 = exp2f(-0.5f * (float)(h + 1)) * LOG2E;

    float m[8], l[8];
#pragma unroll
    for (int v = 0; v < 8; ++v) { m[v] = -1e30f; l[v] = 0.0f; }

    // ================= PASS 1: softmax stats =================
    for (int kc = 0; kc < kS; kc += 32) {
        __syncthreads();
        // stage K chunk 32x64: 32 rows x 8 chunks of 8 halves, async
#pragma unroll
        for (int t = 0; t < 2; ++t) {
            int e = tid + t * 128;               // 0..255
            int row = e >> 3, c = e & 7;
            async_ld16(&sK[row * 72 + c * 8], Kh + (size_t)(kc + row) * 64 + c * 8);
        }
        async_wait0();
        __syncthreads();

        v8f c0, c1;
#pragma unroll
        for (int i = 0; i < 8; ++i) { c0[i] = 0.0f; c1[i] = 0.0f; }
        {
            const _Float16* krow0 = &sK[r * 72];
            const _Float16* krow1 = &sK[(16 + r) * 72];
            v16h b00 = ld_frag(krow0, hi), b01 = ld_frag(krow0 + 32, hi);
            v16h b10 = ld_frag(krow1, hi), b11 = ld_frag(krow1 + 32, hi);
            c0 = WMMA_F16(qa0, b00, c0);
            c0 = WMMA_F16(qa1, b01, c0);
            c1 = WMMA_F16(qa0, b10, c1);
            c1 = WMMA_F16(qa1, b11, c1);
        }
        float key0 = (float)(kc + r);
        float key1 = (float)(kc + 16 + r);
#pragma unroll
        for (int v = 0; v < 8; ++v) {
            float s0 = c0[v] * scale2 + slope2 * key0;     // log2-domain scores
            float s1 = c1[v] * scale2 + slope2 * key1;
            float mn = fmaxf(m[v], fmaxf(s0, s1));
            l[v] = l[v] * EXP2(m[v] - mn) + EXP2(s0 - mn) + EXP2(s1 - mn);
            m[v] = mn;
        }
    }
    // merge 16 partial (m,l) across the lane group (exact LSE merge)
#pragma unroll
    for (int v = 0; v < 8; ++v) {
#pragma unroll
        for (int st = 1; st < 16; st <<= 1) {
            float om = __shfl_xor(m[v], st, 32);
            float ol = __shfl_xor(l[v], st, 32);
            float mn = fmaxf(m[v], om);
            l[v] = l[v] * EXP2(m[v] - mn) + ol * EXP2(om - mn);
            m[v] = mn;
        }
    }
    float rl[8];
#pragma unroll
    for (int v = 0; v < 8; ++v) rl[v] = __builtin_amdgcn_rcpf(l[v]);

    v8f oacc[4];
#pragma unroll
    for (int nt = 0; nt < 4; ++nt)
#pragma unroll
        for (int i = 0; i < 8; ++i) oacc[nt][i] = 0.0f;

    float* wbase = Wout + (((size_t)b * kH + h) * kS + q0 + wave * 16) * (size_t)kS;
    _Float16* wst = &sW[wave * 16 * 40];

    // ================= PASS 2: weights + P*V =================
    for (int kc = 0; kc < kS; kc += 32) {
        __syncthreads();
        // stage K chunk (async)
#pragma unroll
        for (int t = 0; t < 2; ++t) {
            int e = tid + t * 128;
            int row = e >> 3, c = e & 7;
            async_ld16(&sK[row * 72 + c * 8], Kh + (size_t)(kc + row) * 64 + c * 8);
        }
        // stage V chunk transposed: [dv][key]  (needs VGPR transpose, regular path)
#pragma unroll
        for (int t = 0; t < 8; ++t) {
            int e = tid + t * 128;                     // 0..1023 half-pairs
            int key = e >> 5, dv = (e & 31) * 2;
            union { unsigned int u; _Float16 hh[2]; } p;
            p.u = *(const unsigned int*)(const void*)(Vh + (size_t)(kc + key) * 64 + dv);
            sVt[dv * 40 + key]       = p.hh[0];
            sVt[(dv + 1) * 40 + key] = p.hh[1];
        }
        async_wait0();
        __syncthreads();

#pragma unroll
        for (int sub = 0; sub < 2; ++sub) {
            const _Float16* krow = &sK[(sub * 16 + r) * 72];
            v16h b0 = ld_frag(krow, hi);
            v16h b1 = ld_frag(krow + 32, hi);
            v8f c;
#pragma unroll
            for (int i = 0; i < 8; ++i) c[i] = 0.0f;
            c = WMMA_F16(qa0, b0, c);
            c = WMMA_F16(qa1, b1, c);
            float key  = (float)(kc + sub * 16 + r);
            int   kcol = sub * 16 + r;
#pragma unroll
            for (int v = 0; v < 8; ++v) {
                float s = c[v] * scale2 + slope2 * key;
                float w = EXP2(s - m[v]) * rl[v];
                int rrow = v + 8 * hi;
                wbase[(size_t)rrow * kS + kc + kcol] = w;      // weights output
                wst[rrow * 40 + kcol] = (_Float16)w;           // stage for P*V
            }
        }
        // P (16x32) * V (32x64) -> accumulate 16x64
        v16h pa = ld_frag(&wst[r * 40], hi);
#pragma unroll
        for (int nt = 0; nt < 4; ++nt) {
            v16h bv = ld_frag(&sVt[(nt * 16 + r) * 40], hi);
            oacc[nt] = WMMA_F16(pa, bv, oacc[nt]);
        }
    }

    // ---- write context [b, s, h*64 + dv] as f16 ----
#pragma unroll
    for (int nt = 0; nt < 4; ++nt) {
#pragma unroll
        for (int v = 0; v < 8; ++v) {
            int row = q0 + wave * 16 + v + 8 * hi;
            int col = h * 64 + nt * 16 + r;
            ctx[((size_t)b * kS + row) * kD + col] = (_Float16)oacc[nt][v];
        }
    }
}

// ============================================================================
extern "C" void kernel_launch(void* const* d_in, const int* in_sizes, int n_in,
                              void* d_out, int out_size, void* d_ws, size_t ws_size,
                              hipStream_t stream) {
    const float* q  = (const float*)d_in[0];
    const float* k  = (const float*)d_in[1];
    const float* v  = (const float*)d_in[2];
    const float* Wq = (const float*)d_in[3];
    const float* Wk = (const float*)d_in[4];
    const float* Wv = (const float*)d_in[5];
    const float* Wo = (const float*)d_in[6];

    float* out     = (float*)d_out;
    float* weights = out + (size_t)kB * kS * kD;   // output first, then weights

    char* ws = (char*)d_ws;
    const size_t HBUF = (size_t)kB * kS * kD * sizeof(_Float16);  // 8 MB each
    _Float16* Qw  = (_Float16*)(ws + 0 * HBUF);
    _Float16* Kw  = (_Float16*)(ws + 1 * HBUF);
    _Float16* Vw  = (_Float16*)(ws + 2 * HBUF);
    _Float16* ctx = (_Float16*)(ws + 3 * HBUF);

    dim3 gg(kD / 128, (kB * kS) / 128);   // (8, 32)
    dim3 gb(256);

    // QKV projections -> f16 head layout [b,h,s,64]
    gemm_kernel<false, true><<<gg, gb, 0, stream>>>((const void*)q, Wq, (void*)Qw);
    gemm_kernel<false, true><<<gg, gb, 0, stream>>>((const void*)k, Wk, (void*)Kw);
    gemm_kernel<false, true><<<gg, gb, 0, stream>>>((const void*)v, Wv, (void*)Vw);

    // fused attention: weights -> d_out, context -> ws (f16)
    attn_kernel<<<kB * kH * (kS / 64), 128, 0, stream>>>(Qw, Kw, Vw, weights, ctx);

    // output projection: f16 context @ f32 Wo -> f32 d_out
    gemm_kernel<true, false><<<gg, gb, 0, stream>>>((const void*)ctx, Wo, (void*)out);
}
